// ObjectMessage_32427003084983
// MI455X (gfx1250) — matrix operand order repaired and checked by reading, not compiled, not run on hardware
//
#include <hip/hip_runtime.h>
#include <stdint.h>

#define ENABLE_TDM 1   // set to 0 to fall back to manual global->LDS staging

// ---------------------------------------------------------------------------
// Types for CDNA5 WMMA (wave32, V_WMMA_F32_16X16X32_BF16)
// ---------------------------------------------------------------------------
typedef __attribute__((ext_vector_type(16))) __bf16    v16bf;
typedef __attribute__((ext_vector_type(8)))  float     v8f;
typedef __attribute__((ext_vector_type(4)))  uint32_t  u32x4;
typedef __attribute__((ext_vector_type(8)))  uint32_t  u32x8;

struct alignas(16) U16B { uint32_t u[4]; };

static __device__ __forceinline__ __bf16 f2bf(float f) {
  union { float f; uint32_t u; } cu; cu.f = f;
  uint32_t u = cu.u;
  uint32_t r = (u + 0x7FFFu + ((u >> 16) & 1u)) >> 16;  // round-to-nearest-even
  union { unsigned short s; __bf16 b; } cb; cb.s = (unsigned short)r;
  return cb.b;
}

#if ENABLE_TDM
// Issue a TDM copy of a 128-row x 32-half tile (row-major, element=2B) from
// global memory into LDS with 4-DWORD padding after every 16 DWORDs (so rows
// land on the 40-half LDS pitch). Per CDNA5 ISA ch.7/8: D# group0 holds
// count/lds_addr/global_addr/type=2, group1 holds sizes/strides/pad config.
// Issued by one wave; completion via TENSORcnt, published by the WG barrier.
static __device__ __forceinline__ void tdm_load_tile_128x32(
    const __bf16* gsrc, int ldSrc, void* ldsDst) {
  uint64_t ga = (uint64_t)(uintptr_t)gsrc;
  u32x4 g0;
  g0[0] = 1u;                                          // count=1, user mode
  g0[1] = (uint32_t)(uintptr_t)ldsDst;                 // LDS byte address (addr[31:0])
  g0[2] = (uint32_t)ga;                                // global_addr[31:0]
  g0[3] = ((uint32_t)(ga >> 32) & 0x01FFFFFFu)         // global_addr[56:32]
        | (2u << 30);                                  // type = 2 ("image")
  u32x8 g1;
  g1[0] = (1u << 16)                                   // data_size = 2 bytes
        | (1u << 20)                                   // pad_enable
        | (3u << 22)                                   // pad_interval: 16 DWORDs
        | (3u << 25);                                  // pad_amount: 4 DWORDs
  g1[1] = (32u << 16);                                 // tensor_dim0 = 32 (lo16)
  g1[2] = (0u) | (128u << 16);                         // td0 hi16=0 | tensor_dim1 lo16=128
  g1[3] = (0u) | (32u << 16);                          // td1 hi16=0 | tile_dim0=32
  g1[4] = 128u;                                        // tile_dim1=128, tile_dim2=0
  g1[5] = (uint32_t)ldSrc;                             // tensor_dim0_stride[31:0]
  g1[6] = 0u;                                          // stride hi / dim1_stride lo
  g1[7] = 0u;
  asm volatile("tensor_load_to_lds %0, %1" :: "s"(g0), "s"(g1) : "memory");
}
#endif

// ---------------------------------------------------------------------------
// Generic bf16 WMMA GEMM: C = act(A[M,K] @ B[K,N]) (+residual) per batch z.
//  - 256 threads = 8 waves; block tile 128x128, wave tile 32x64 (2x4 WMMA)
//  - BK = 32; LDS pitch 40 halves (16B-aligned fragments, conflict-reduced)
//  - Full tiles staged by the Tensor Data Mover; K-tails via guarded loads
//  - TRANSB: B[k][n] = Bsrc[n*ldb + k] (u1 @ u1^T); TDM-stageable (row copy)
// ---------------------------------------------------------------------------
template<bool RELU, bool TRANSB, bool OUT_F32, bool OUT_BF16, bool RESID>
__global__ __launch_bounds__(256)
void gemm_wmma_bf16(const __bf16* __restrict__ A, long bsA, int lda,
                    const __bf16* __restrict__ Bm, long bsB, int ldb,
                    float* __restrict__ Cf, __bf16* __restrict__ Cb,
                    long bsC, int ldc,
                    const float* __restrict__ Res,
                    int M, int N, int K)
{
  constexpr int LDSP = 40;                 // LDS row pitch in halves (80B)
  __shared__ __bf16 As[128][LDSP];         // [m][k]
  __shared__ __bf16 Bs[128][LDSP];         // [n][k]

  const int bz = blockIdx.z;
  A  += bsA * (long)bz;
  Bm += bsB * (long)bz;
  const size_t cBase = (size_t)(bsC * (long)bz);

  const int mTile = blockIdx.y * 128;
  const int nTile = blockIdx.x * 128;
  const int tid   = threadIdx.x;
  const int lane  = tid & 31;
  const int wid   = tid >> 5;
  const int waveM = wid >> 1;              // 0..3 -> 32 rows each
  const int waveN = wid & 1;               // 0..1 -> 64 cols each
  const int l16   = lane & 15;
  const int lhi   = lane >> 4;             // 0 or 1

  v8f acc[2][4];
  {
    v8f z; for (int r = 0; r < 8; ++r) z[r] = 0.f;
    for (int a = 0; a < 2; ++a) for (int b = 0; b < 4; ++b) acc[a][b] = z;
  }

  for (int k0 = 0; k0 < K; k0 += 32) {
    const bool fullK = (k0 + 32 <= K);
#if ENABLE_TDM
    const bool tdmA = fullK && (mTile + 128 <= M);
    const bool tdmB = TRANSB && fullK && (nTile + 128 <= N);
    if ((tdmA || tdmB) && wid == 0) {
      if (tdmA) tdm_load_tile_128x32(A  + (size_t)mTile * lda + k0, lda, &As[0][0]);
      if (tdmB) tdm_load_tile_128x32(Bm + (size_t)nTile * ldb + k0, ldb, &Bs[0][0]);
      __builtin_amdgcn_s_wait_tensorcnt(0);
    }
#else
    const bool tdmA = false, tdmB = false;
#endif
    // ---- stage A tile: 128 x 32 halves, 16B per thread-load ----
    if (!tdmA) {
      #pragma unroll
      for (int r = 0; r < 2; ++r) {
        int i   = tid + r * 256;
        int row = i >> 2;
        int kc  = (i & 3) << 3;              // 0,8,16,24
        int gm  = mTile + row;
        int gk  = k0 + kc;
        U16B val; val.u[0] = val.u[1] = val.u[2] = val.u[3] = 0;
        if (gm < M) {
          if (gk + 7 < K) {
            val = *(const U16B*)(A + (size_t)gm * lda + gk);
          } else {
            alignas(16) __bf16 tmp[8] = {};
            for (int j = 0; j < 8; ++j) if (gk + j < K) tmp[j] = A[(size_t)gm * lda + gk + j];
            val = *(const U16B*)tmp;
          }
        }
        *(U16B*)&As[row][kc] = val;
      }
    }
    // ---- stage B tile into [n][k] layout ----
    if (TRANSB) {
      if (!tdmB) {
        #pragma unroll
        for (int r = 0; r < 2; ++r) {
          int i  = tid + r * 256;
          int n  = i >> 2;
          int kc = (i & 3) << 3;
          int gn = nTile + n;
          int gk = k0 + kc;
          U16B val; val.u[0] = val.u[1] = val.u[2] = val.u[3] = 0;
          if (gn < N) {
            if (gk + 7 < K) {
              val = *(const U16B*)(Bm + (size_t)gn * ldb + gk);
            } else {
              alignas(16) __bf16 tmp[8] = {};
              for (int j = 0; j < 8; ++j) if (gk + j < K) tmp[j] = Bm[(size_t)gn * ldb + gk + j];
              val = *(const U16B*)tmp;
            }
          }
          *(U16B*)&Bs[n][kc] = val;
        }
      }
    } else {
      // transpose staging (TDM cannot transpose): [k][n] global -> [n][k] LDS
      #pragma unroll
      for (int r = 0; r < 2; ++r) {
        int i   = tid + r * 256;
        int k   = i >> 4;                  // 0..31
        int nc  = (i & 15) << 3;           // 0..120
        int gk  = k0 + k;
        int gn0 = nTile + nc;
        alignas(16) __bf16 tmp[8] = {};
        if (gk < K) {
          if (gn0 + 7 < N) {
            *(U16B*)tmp = *(const U16B*)(Bm + (size_t)gk * ldb + gn0);
            if (k0 + 32 < K) __builtin_prefetch(Bm + (size_t)(gk + 32) * ldb + gn0, 0, 1);
          } else {
            for (int j = 0; j < 8; ++j) if (gn0 + j < N) tmp[j] = Bm[(size_t)gk * ldb + gn0 + j];
          }
        }
        #pragma unroll
        for (int j = 0; j < 8; ++j) Bs[nc + j][k] = tmp[j];
      }
    }
    __syncthreads();

    // ---- fragments per ISA 7.12.2 16-bit layouts ----
    union Frag { v16bf v; U16B q[2]; };
    Frag fa[2], fb[4];
    #pragma unroll
    for (int mf = 0; mf < 2; ++mf) {
      int row = waveM * 32 + mf * 16 + l16;
      int kb  = lhi ? 8 : 0;               // lanes>=16 hold K 8..15 / 24..31
      fa[mf].q[0] = *(const U16B*)&As[row][kb];
      fa[mf].q[1] = *(const U16B*)&As[row][kb + 16];
    }
    #pragma unroll
    for (int nf = 0; nf < 4; ++nf) {
      int n  = waveN * 64 + nf * 16 + l16;
      int kb = lhi ? 8 : 0;                // lanes>=16 hold K 8..15 / 24..31
      fb[nf].q[0] = *(const U16B*)&Bs[n][kb];
      fb[nf].q[1] = *(const U16B*)&Bs[n][kb + 16];
    }
    #pragma unroll
    for (int mf = 0; mf < 2; ++mf)
      #pragma unroll
      for (int nf = 0; nf < 4; ++nf)
        acc[mf][nf] = __builtin_amdgcn_wmma_f32_16x16x32_bf16(
            false, fa[mf].v, false, fb[nf].v, (short)0, acc[mf][nf], false, false);
    __syncthreads();
  }

  // ---- epilogue: C layout M = r + 8*lhi, N = l16 ----
  #pragma unroll
  for (int mf = 0; mf < 2; ++mf)
    #pragma unroll
    for (int nf = 0; nf < 4; ++nf)
      #pragma unroll
      for (int r = 0; r < 8; ++r) {
        int row = mTile + waveM * 32 + mf * 16 + lhi * 8 + r;
        int col = nTile + waveN * 64 + nf * 16 + l16;
        if (row < M && col < N) {
          float v = acc[mf][nf][r];
          if (RELU) v = fmaxf(v, 0.f);
          size_t off = cBase + (size_t)row * ldc + col;
          if (RESID) v += Res[off];
          if (OUT_F32) Cf[off] = v;
          if (OUT_BF16) Cb[off] = f2bf(v);
        }
      }
}

// ---------------------------------------------------------------------------
// Auxiliary kernels
// ---------------------------------------------------------------------------
__global__ void cvt_bf16(const float* __restrict__ src, __bf16* __restrict__ dst,
                         int rows, int cols, int ldd) {
  long i = (long)blockIdx.x * 256 + threadIdx.x;
  long tot = (long)rows * cols;
  if (i >= tot) return;
  int r = (int)(i / cols), c = (int)(i % cols);
  dst[(size_t)r * ldd + c] = f2bf(src[i]);
}

__global__ __launch_bounds__(256)
void rowinv_norm(const float* __restrict__ u1, float* __restrict__ rinv,
                 int cols, int ld) {
  int row = blockIdx.x;
  __shared__ float red[256];
  float s = 0.f;
  for (int c = threadIdx.x; c < cols; c += 256) {
    float v = u1[(size_t)row * ld + c];
    s += v * v;
  }
  red[threadIdx.x] = s; __syncthreads();
  for (int o = 128; o; o >>= 1) {
    if ((int)threadIdx.x < o) red[threadIdx.x] += red[threadIdx.x + o];
    __syncthreads();
  }
  if (threadIdx.x == 0) rinv[row] = rsqrtf(red[0]);
}

__global__ void sim_finalize(float* __restrict__ S, const float* __restrict__ rinv) {
  int i = blockIdx.x * 256 + threadIdx.x;
  if (i >= 16 * 128 * 128) return;
  int b = i >> 14, n = (i >> 7) & 127, m = i & 127;
  float v = S[i] * rinv[b * 128 + n] * rinv[b * 128 + m];
  S[i] = fminf(1.f, fmaxf(-1.f, v));
}

__global__ __launch_bounds__(256)
void argmax151(const float* __restrict__ dists, int* __restrict__ preds) {
  int wid = threadIdx.x >> 5, lane = threadIdx.x & 31;
  int row = blockIdx.x * 8 + wid;
  const float* d = dists + (size_t)row * 151;
  float bv = -__builtin_inff(); int bi = 0x7fffffff;
  for (int c = lane; c < 151; c += 32) {
    float v = d[c];
    if (v > bv || (v == bv && c < bi)) { bv = v; bi = c; }
  }
  for (int off = 16; off; off >>= 1) {
    float ov = __shfl_xor(bv, off);
    int   oi = __shfl_xor(bi, off);
    if (ov > bv || (ov == bv && oi < bi)) { bv = ov; bi = oi; }
  }
  if (lane == 0) preds[row] = bi;
}

__global__ __launch_bounds__(256)
void top2_kernel(const float* __restrict__ sim, int* __restrict__ idx2) {
  int wid = threadIdx.x >> 5, lane = threadIdx.x & 31;
  int row = blockIdx.x * 8 + wid;           // (b*128 + n)
  const float* s = sim + (size_t)row * 128;
  float v1 = -__builtin_inff(), v2 = -__builtin_inff();
  int i1 = 0x7fffffff, i2 = 0x7fffffff;
  for (int j = 0; j < 4; ++j) {
    int m = lane * 4 + j;
    float v = s[m];
    if (v > v1 || (v == v1 && m < i1)) { v2 = v1; i2 = i1; v1 = v; i1 = m; }
    else if (v > v2 || (v == v2 && m < i2)) { v2 = v; i2 = m; }
  }
  for (int off = 16; off; off >>= 1) {
    float ov1 = __shfl_xor(v1, off), ov2 = __shfl_xor(v2, off);
    int   oi1 = __shfl_xor(i1, off), oi2 = __shfl_xor(i2, off);
    if (ov1 > v1 || (ov1 == v1 && oi1 < i1)) {
      float nv2; int ni2;
      if (v1 > ov2 || (v1 == ov2 && i1 < oi2)) { nv2 = v1; ni2 = i1; }
      else { nv2 = ov2; ni2 = oi2; }
      v1 = ov1; i1 = oi1; v2 = nv2; i2 = ni2;
    } else if (ov1 > v2 || (ov1 == v2 && oi1 < i2)) {
      v2 = ov1; i2 = oi1;
    }
  }
  if (lane == 0) { idx2[row * 2] = i1; idx2[row * 2 + 1] = i2; }
}

__global__ __launch_bounds__(256)
void gather_kernel(const float* __restrict__ u1, const int* __restrict__ idx2,
                   __bf16* __restrict__ gath, int ldu, int ldg) {
  int row = blockIdx.x;                      // b*128 + m
  int b   = row >> 7;
  int i0  = idx2[row * 2], i1 = idx2[row * 2 + 1];
  const float* s0 = u1 + (size_t)(b * 128 + i0) * ldu;
  const float* s1 = u1 + (size_t)(b * 128 + i1) * ldu;
  __bf16* g = gath + (size_t)row * ldg;
  for (int c = threadIdx.x; c < 1106; c += 256) {
    g[c]        = f2bf(s0[c]);
    g[1106 + c] = f2bf(s1[c]);
  }
}

// conv1: in = [freq(51) | sim] 52ch built on the fly; 3x3 SAME, 52 -> 10
__global__ __launch_bounds__(256)
void conv1_kernel(const float* __restrict__ freq, const float* __restrict__ sim,
                  const int* __restrict__ preds, const float* __restrict__ W1,
                  float* __restrict__ h1) {
  int i = blockIdx.x * 256 + threadIdx.x;
  if (i >= 16 * 10 * 128 * 128) return;
  int x = i & 127, y = (i >> 7) & 127, o = (i >> 14) % 10, b = i / (10 * 16384);
  float acc = 0.f;
  for (int dy = 0; dy < 3; ++dy) {
    int yy = y + dy - 1; if ((unsigned)yy > 127u) continue;
    int pn = preds[b * 128 + yy];
    for (int dx = 0; dx < 3; ++dx) {
      int xx = x + dx - 1; if ((unsigned)xx > 127u) continue;
      int pm = preds[b * 128 + xx];
      const float* row = freq + (size_t)(pn * 151 + pm) * 51;
      const float* w   = W1 + (size_t)o * 52 * 9 + (dy * 3 + dx);
      float a = 0.f;
      #pragma unroll 3
      for (int c = 0; c < 51; ++c) a += w[c * 9] * row[c];
      a += w[51 * 9] * sim[b * 16384 + yy * 128 + xx];
      acc += a;
    }
  }
  h1[i] = acc;
}

// conv2: 3x3 SAME, 10 -> 5
__global__ __launch_bounds__(256)
void conv2_kernel(const float* __restrict__ h1, const float* __restrict__ W2,
                  float* __restrict__ h2) {
  int i = blockIdx.x * 256 + threadIdx.x;
  if (i >= 16 * 5 * 128 * 128) return;
  int x = i & 127, y = (i >> 7) & 127, o = (i >> 14) % 5, b = i / (5 * 16384);
  float acc = 0.f;
  for (int dy = 0; dy < 3; ++dy) {
    int yy = y + dy - 1; if ((unsigned)yy > 127u) continue;
    for (int dx = 0; dx < 3; ++dx) {
      int xx = x + dx - 1; if ((unsigned)xx > 127u) continue;
      #pragma unroll
      for (int c = 0; c < 10; ++c)
        acc += W2[((o * 10 + c) * 3 + dy) * 3 + dx] *
               h1[((size_t)(b * 10 + c) * 128 + yy) * 128 + xx];
    }
  }
  h2[i] = acc;
}

// conv3: 1x1, 5 -> 1, then tanh, emit bf16 adjacency
__global__ void conv3_kernel(const float* __restrict__ h2, const float* __restrict__ W3,
                             __bf16* __restrict__ adj) {
  int i = blockIdx.x * 256 + threadIdx.x;
  if (i >= 16 * 128 * 128) return;
  int b = i >> 14, yx = i & 16383;
  float acc = 0.f;
  #pragma unroll
  for (int c = 0; c < 5; ++c) acc += W3[c] * h2[(size_t)(b * 5 + c) * 16384 + yx];
  adj[i] = f2bf(tanhf(acc));
}

// ---------------------------------------------------------------------------
// Launch
// ---------------------------------------------------------------------------
extern "C" void kernel_launch(void* const* d_in, const int* in_sizes, int n_in,
                              void* d_out, int out_size, void* d_ws, size_t ws_size,
                              hipStream_t stream) {
  (void)in_sizes; (void)n_in; (void)out_size; (void)ws_size;

  const float* feat  = (const float*)d_in[0];   // [2048][4424]
  const float* dists = (const float*)d_in[1];   // [2048][151]
  const float* Wcomp = (const float*)d_in[2];   // [4424][1106]
  const float* Wou1  = (const float*)d_in[3];   // [1106][1106]
  const float* Wofc  = (const float*)d_in[4];   // [2212][1106]
  const float* Wdec  = (const float*)d_in[5];   // [1106][4424]
  const float* W1    = (const float*)d_in[6];   // [10][52][3][3]
  const float* W2    = (const float*)d_in[7];   // [5][10][3][3]
  const float* W3    = (const float*)d_in[8];   // [1][5][1][1]
  const float* freq  = (const float*)d_in[9];   // [151*151][51]

  const int Bn = 16, Nn = 128, D = 4424, d4 = 1106;
  const int R   = Bn * Nn;                      // 2048 rows
  const int d4p = 1112;                         // padded ld (16B-aligned bf16 rows)
  const int d2p = 2216;                         // padded 2*d4

  char* p = (char*)d_ws;
  auto alloc = [&](size_t bytes) {
    char* r = p; p += (bytes + 255) & ~(size_t)255; return r;
  };

  __bf16* feat_bf  = (__bf16*)alloc((size_t)R * D * 2);
  __bf16* Wcomp_bf = (__bf16*)alloc((size_t)D * d4p * 2);
  __bf16* Wou1_bf  = (__bf16*)alloc((size_t)d4 * d4p * 2);
  __bf16* Wofc_bf  = (__bf16*)alloc((size_t)(2 * d4) * d4p * 2);
  __bf16* Wdec_bf  = (__bf16*)alloc((size_t)d4 * D * 2);
  __bf16* comp_bf  = (__bf16*)alloc((size_t)R * d4p * 2);
  float*  u1f      = (float*) alloc((size_t)R * d4p * 4);
  __bf16* u1_bf    = (__bf16*)alloc((size_t)R * d4p * 2);
  float*  rinv     = (float*) alloc((size_t)R * 4);
  float*  S        = (float*) alloc((size_t)Bn * Nn * Nn * 4);
  int*    preds    = (int*)   alloc((size_t)R * 4);
  int*    idx2     = (int*)   alloc((size_t)R * 2 * 4);
  __bf16* gath_bf  = (__bf16*)alloc((size_t)R * d2p * 2);
  float*  h1       = (float*) alloc((size_t)Bn * 10 * Nn * Nn * 4);
  float*  h2       = (float*) alloc((size_t)Bn * 5 * Nn * Nn * 4);
  __bf16* adj_bf   = (__bf16*)alloc((size_t)Bn * Nn * Nn * 2);
  __bf16* ofl_bf   = (__bf16*)alloc((size_t)R * d2p * 2);
  __bf16* um_bf    = (__bf16*)alloc((size_t)R * d4p * 2);

  auto cvtGrid = [](long tot) { return (unsigned)((tot + 255) / 256); };

  // f32 -> bf16 (padded leading dims keep all 16B vector loads aligned)
  cvt_bf16<<<cvtGrid((long)R * D),  256, 0, stream>>>(feat,  feat_bf,  R,      D,  D);
  cvt_bf16<<<cvtGrid((long)D * d4), 256, 0, stream>>>(Wcomp, Wcomp_bf, D,      d4, d4p);
  cvt_bf16<<<cvtGrid((long)d4 * d4),256, 0, stream>>>(Wou1,  Wou1_bf,  d4,     d4, d4p);
  cvt_bf16<<<cvtGrid((long)2*d4*d4),256, 0, stream>>>(Wofc,  Wofc_bf,  2 * d4, d4, d4p);
  cvt_bf16<<<cvtGrid((long)d4 * D), 256, 0, stream>>>(Wdec,  Wdec_bf,  d4,     D,  D);

  // G1: comp = relu(feat @ W_comp)            [2048 x 1106], K = 4424
  gemm_wmma_bf16<true, false, false, true, false>
      <<<dim3((d4 + 127) / 128, (R + 127) / 128, 1), 256, 0, stream>>>(
      feat_bf, 0, D, Wcomp_bf, 0, d4p, nullptr, comp_bf, 0, d4p, nullptr, R, d4, D);

  // G2: u1 = relu(comp @ W_ou1)               f32 + bf16 outputs
  gemm_wmma_bf16<true, false, true, true, false>
      <<<dim3((d4 + 127) / 128, (R + 127) / 128, 1), 256, 0, stream>>>(
      comp_bf, 0, d4p, Wou1_bf, 0, d4p, u1f, u1_bf, 0, d4p, nullptr, R, d4, d4);

  // row inverse norms
  rowinv_norm<<<R, 256, 0, stream>>>(u1f, rinv, d4, d4p);

  // G3: S[b] = u1[b] @ u1[b]^T  (TRANSB path) [16 x 128 x 128], K = 1106
  gemm_wmma_bf16<false, true, true, false, false>
      <<<dim3(1, 1, Bn), 256, 0, stream>>>(
      u1_bf, (long)Nn * d4p, d4p, u1_bf, (long)Nn * d4p, d4p,
      S, nullptr, (long)Nn * Nn, Nn, nullptr, Nn, Nn, d4);

  // sim = clip(S * rinv_n * rinv_m, -1, 1)
  sim_finalize<<<cvtGrid((long)Bn * Nn * Nn), 256, 0, stream>>>(S, rinv);

  // obj predictions + top-2 neighbors + gather
  argmax151<<<R / 8, 256, 0, stream>>>(dists, preds);
  top2_kernel<<<R / 8, 256, 0, stream>>>(S, idx2);
  gather_kernel<<<R, 256, 0, stream>>>(u1f, idx2, gath_bf, d4p, d2p);

  // conv stack (freq-table rows fetched on the fly; L2-resident)
  conv1_kernel<<<cvtGrid((long)Bn * 10 * Nn * Nn), 256, 0, stream>>>(freq, S, preds, W1, h1);
  conv2_kernel<<<cvtGrid((long)Bn * 5  * Nn * Nn), 256, 0, stream>>>(h1, W2, h2);
  conv3_kernel<<<cvtGrid((long)Bn * Nn * Nn),      256, 0, stream>>>(h2, W3, adj_bf);

  // G4: ofl[b] = adj[b] @ gathered[b]         [16 x 128 x 2212], K = 128
  gemm_wmma_bf16<false, false, false, true, false>
      <<<dim3((2 * d4 + 127) / 128, 1, Bn), 256, 0, stream>>>(
      adj_bf, (long)Nn * Nn, Nn, gath_bf, (long)Nn * d2p, d2p,
      nullptr, ofl_bf, (long)Nn * d2p, d2p, nullptr, Nn, 2 * d4, Nn);

  // G5: u_m = relu(ofl @ W_ofc)               [2048 x 1106], K = 2212
  gemm_wmma_bf16<true, false, false, true, false>
      <<<dim3((d4 + 127) / 128, (R + 127) / 128, 1), 256, 0, stream>>>(
      ofl_bf, 0, d2p, Wofc_bf, 0, d4p, nullptr, um_bf, 0, d4p, nullptr, R, d4, 2 * d4);

  // G6: out = relu(u_m @ W_decomp) + feat     [2048 x 4424], K = 1106
  gemm_wmma_bf16<true, false, true, false, true>
      <<<dim3((D + 127) / 128, (R + 127) / 128, 1), 256, 0, stream>>>(
      um_bf, 0, d4p, Wdec_bf, 0, D, (float*)d_out, nullptr, 0, D, feat, R, D, d4);
}
